// CACISLoss_72361609003425
// MI455X (gfx1250) — matrix operand order repaired and checked by reading, not compiled, not run on hardware
//
#include <hip/hip_runtime.h>
#include <hip/hip_bf16.h>
#include <math.h>

// ---------------------------------------------------------------------------
// CACIS Fenchel-Young loss, CDNA5 (gfx1250), wave32.
//   B = 4096 examples, K = 128.
//   One workgroup (256 threads = 8 waves) per example.
//   Single pass over C (HBM-bound: ~11us floor), M kept as f16 in LDS,
//   Frank-Wolfe mat-vec via v_wmma_f32_16x16x32_f16 with the A operand
//   (rows of M) hoisted into registers for all 50 iterations.
// ---------------------------------------------------------------------------

typedef _Float16 v16h __attribute__((ext_vector_type(16)));
typedef _Float16 v8h  __attribute__((ext_vector_type(8)));
typedef _Float16 v4h  __attribute__((ext_vector_type(4)));
typedef float    v8f  __attribute__((ext_vector_type(8)));

#define K_DIM       128
#define SOLVER_ITER 50
#define EPS_SCALE   1.0f
#define EPS_MIN     1e-08f

// ---- dynamic LDS layout (bytes) -------------------------------------------
//  Tbuf  : 16384 f32  (staged t_ij = f_i + f_j + C_ij)        65536
//  Mbuf  : 16384 f16  (M = exp((min - t)/eps))                32768
//  sf    :   128 f32  (scores row)                              512
//  alpha :   128 f32                                            512
//  grad  :   128 f32                                            512
//  rv    :   256 f32  (reduction scratch)                      1024
//  ah    :   128 f16  (alpha in f16 for WMMA B operand)          256
#define OFF_T     0
#define OFF_M     65536
#define OFF_SF    (OFF_M + 32768)
#define OFF_ALPHA (OFF_SF + 512)
#define OFF_GRAD  (OFF_ALPHA + 512)
#define OFF_RV    (OFF_GRAD + 512)
#define OFF_AH    (OFF_RV + 1024)
#define LDS_BYTES (OFF_AH + 256 + 128)   // pad

// grad[0..127] = M @ alpha with A tiles already resident in registers.
// Wave w owns rows [16w, 16w+16); A[kb] covers K-block kb (32 wide).
// B operand carries alpha broadcast across all 16 columns; column 0 of D
// (held by lanes 0 and 16) is the result vector block.
__device__ __forceinline__ void matvec_wmma_reg(const v16h (&A)[4],
                                                const _Float16* __restrict__ alpha_h,
                                                float* __restrict__ grad_out,
                                                int tid) {
  const int wave = tid >> 5;
  const int lane = tid & 31;
  const int hi   = lane >> 4;

  v8f acc = {};
#pragma unroll
  for (int kb = 0; kb < 4; ++kb) {
    const int ck = kb * 32;
    // B tile: 32x16, elems 0-7 K = ck + 8*hi + [0..7], elems 8-15 K = ck + 16 + 8*hi + [0..7].
    // alpha broadcast across columns -> two 8-half chunks per lane.
    v16h bm;
    *((v8h*)&bm)     = *(const v8h*)(alpha_h + ck + 8 * hi);
    *((v8h*)&bm + 1) = *(const v8h*)(alpha_h + ck + 16 + 8 * hi);

    acc = __builtin_amdgcn_wmma_f32_16x16x32_f16(
        /*neg_a=*/false, A[kb], /*neg_b=*/false, bm,
        /*c_mod=*/(short)0, acc, /*reuse_a=*/false, /*reuse_b=*/false);
  }
  // D layout (f32 16x16): VGPR g, lane L -> row = g + 8*(L>=16), col = L&15.
  if ((lane & 15) == 0) {
    const int base = (wave << 4) + hi * 8;
#pragma unroll
    for (int g = 0; g < 8; ++g) grad_out[base + g] = acc[g];
  }
}

__global__ void __launch_bounds__(256)
cacis_main_kernel(const float* __restrict__ scores,
                  const float* __restrict__ C,
                  const int*   __restrict__ targets,
                  float*       __restrict__ ws) {
  extern __shared__ __align__(128) char smem[];
  float*    Tbuf    = (float*)(smem + OFF_T);
  _Float16* Mbuf    = (_Float16*)(smem + OFF_M);
  float*    sf      = (float*)(smem + OFF_SF);
  float*    alpha   = (float*)(smem + OFF_ALPHA);
  float*    grad    = (float*)(smem + OFF_GRAD);
  float*    rv      = (float*)(smem + OFF_RV);
  _Float16* alpha_h = (_Float16*)(smem + OFF_AH);

  const int tid = threadIdx.x;
  const int b   = blockIdx.x;
  const float4* C4 = (const float4*)(C + (size_t)b * (K_DIM * K_DIM));

  // scores row into LDS
  if (tid < K_DIM) sf[tid] = scores[(size_t)b * K_DIM + tid];
  __syncthreads();

  // ---- pass 1: stream C once; accumulate sum, trace, min(t); stage t ------
  float psum = 0.0f, ptr = 0.0f, pmin = 3.4e38f;
#pragma unroll
  for (int j = 0; j < 16; ++j) {
    const int q4 = tid + 256 * j;            // float4 index, coalesced
    if (j + 1 < 16)
      __builtin_prefetch((const void*)(C4 + tid + 256 * (j + 1)), 0, 3);
    const float4 c = C4[q4];
    const int q   = q4 * 4;
    const int i   = q >> 7;                  // row
    const int col = q & (K_DIM - 1);         // first of 4 columns
    const float fi = sf[i];
    const float t0 = fi + sf[col + 0] + c.x;
    const float t1 = fi + sf[col + 1] + c.y;
    const float t2 = fi + sf[col + 2] + c.z;
    const float t3 = fi + sf[col + 3] + c.w;
    psum += (c.x + c.y) + (c.z + c.w);
    const int dx = i - col;                  // diagonal element if 0..3
    if (dx >= 0 && dx < 4)
      ptr += (dx == 0) ? c.x : (dx == 1) ? c.y : (dx == 2) ? c.z : c.w;
    pmin = fminf(pmin, fminf(fminf(t0, t1), fminf(t2, t3)));
    float4 tv; tv.x = t0; tv.y = t1; tv.z = t2; tv.w = t3;
    ((float4*)Tbuf)[q4] = tv;
  }

  // ---- block reductions: sum, trace, min ----------------------------------
  rv[tid] = psum; __syncthreads();
  for (int s = 128; s > 0; s >>= 1) { if (tid < s) rv[tid] += rv[tid + s]; __syncthreads(); }
  const float totsum = rv[0]; __syncthreads();

  rv[tid] = ptr; __syncthreads();
  for (int s = 128; s > 0; s >>= 1) { if (tid < s) rv[tid] += rv[tid + s]; __syncthreads(); }
  const float tottr = rv[0]; __syncthreads();

  rv[tid] = pmin; __syncthreads();
  for (int s = 128; s > 0; s >>= 1) { if (tid < s) rv[tid] = fminf(rv[tid], rv[tid + s]); __syncthreads(); }
  const float minv = rv[0]; __syncthreads();

  const float eps = fmaxf(EPS_SCALE * (totsum - tottr) /
                          ((float)K_DIM * (float)(K_DIM - 1)), EPS_MIN);
  const float inv_eps = 1.0f / eps;
  // shift = -minv/eps  =>  exponent - shift = (minv - t)/eps  (<= 0)

  // ---- pass 2: M = exp((minv - t)/eps) as f16 in LDS ----------------------
#pragma unroll
  for (int j = 0; j < 16; ++j) {
    const int q4 = tid + 256 * j;
    const float4 tv = ((const float4*)Tbuf)[q4];
    v4h m;
    m[0] = (_Float16)__expf((minv - tv.x) * inv_eps);
    m[1] = (_Float16)__expf((minv - tv.y) * inv_eps);
    m[2] = (_Float16)__expf((minv - tv.z) * inv_eps);
    m[3] = (_Float16)__expf((minv - tv.w) * inv_eps);
    *(v4h*)(Mbuf + q4 * 4) = m;
  }
  if (tid < K_DIM) {
    const float a0 = 1.0f / (float)K_DIM;
    alpha[tid]   = a0;
    alpha_h[tid] = (_Float16)a0;
  }
  __syncthreads();                           // Mbuf + alpha ready

  // ---- hoist A operand (rows of M) into registers: invariant over FW ------
  // A tile (16-bit 16x32 layout): lane&15 = M row within wave's 16-row block;
  //  VGPR0-3: K = ck + 8*hi + [0..7]   VGPR4-7: K = ck + 16 + 8*hi + [0..7]
  const int wave = tid >> 5;
  const int lane = tid & 31;
  const int hi   = lane >> 4;
  const _Float16* Mrow = Mbuf + ((wave << 4) + (lane & 15)) * K_DIM;
  v16h A[4];
#pragma unroll
  for (int kb = 0; kb < 4; ++kb) {
    const int ck = kb * 32;
    *((v8h*)&A[kb])     = *(const v8h*)(Mrow + ck + 8 * hi);
    *((v8h*)&A[kb] + 1) = *(const v8h*)(Mrow + ck + 16 + 8 * hi);
  }

  // ---- Frank-Wolfe: 50 iterations -----------------------------------------
  for (int it = 0; it < SOLVER_ITER; ++it) {
    matvec_wmma_reg(A, alpha_h, grad, tid);  // grad = M @ alpha (x2 irrelevant)
    __syncthreads();                         // grad visible
    if (tid < 32) {                          // wave 0: argmin + simplex update
      float bv = grad[tid];
      int   bi = tid;
#pragma unroll
      for (int k = 1; k < 4; ++k) {
        const float v = grad[tid + 32 * k];
        if (v < bv) { bv = v; bi = tid + 32 * k; }
      }
#pragma unroll
      for (int off = 16; off > 0; off >>= 1) {
        const float ov = __shfl_down(bv, off, 32);
        const int   oi = __shfl_down(bi, off, 32);
        if (ov < bv || (ov == bv && oi < bi)) { bv = ov; bi = oi; }
      }
      const int idx = __shfl(bi, 0, 32);     // broadcast winner (first-min)
      const float step = 2.0f / ((float)it + 2.0f);
      const float keep = 1.0f - step;
#pragma unroll
      for (int k = 0; k < 4; ++k) {
        const int j = tid + 32 * k;
        const float a = alpha[j] * keep + ((j == idx) ? step : 0.0f);
        alpha[j]   = a;
        alpha_h[j] = (_Float16)a;
      }
    }
    __syncthreads();                         // alpha update visible
  }

  // ---- val = alpha^T M alpha ----------------------------------------------
  matvec_wmma_reg(A, alpha_h, grad, tid);    // grad = M @ alpha
  __syncthreads();
  if (tid < K_DIM) rv[tid] = alpha[tid] * grad[tid];
  __syncthreads();
  for (int s = 64; s > 0; s >>= 1) { if (tid < s) rv[tid] += rv[tid + s]; __syncthreads(); }

  if (tid == 0) {
    const float val = rv[0];
    // conjugate = -eps*(log(val+1e-12) + shift); -eps*shift = minv
    const float conj = -eps * logf(val + 1e-12f) + minv;
    const int tgt = targets[b];
    ws[b] = conj - sf[tgt];
  }
}

__global__ void __launch_bounds__(256)
cacis_reduce_kernel(const float* __restrict__ ws, float* __restrict__ out, int B) {
  __shared__ float s[256];
  const int t = threadIdx.x;
  float acc = 0.0f;
  for (int i = t; i < B; i += 256) acc += ws[i];
  s[t] = acc; __syncthreads();
  for (int k = 128; k > 0; k >>= 1) { if (t < k) s[t] += s[t + k]; __syncthreads(); }
  if (t == 0) out[0] = s[0] / (float)B;
}

extern "C" void kernel_launch(void* const* d_in, const int* in_sizes, int n_in,
                              void* d_out, int out_size, void* d_ws, size_t ws_size,
                              hipStream_t stream) {
  const float* scores  = (const float*)d_in[0];   // (B,K) f32
  const float* C       = (const float*)d_in[1];   // (B,K,K) f32
  const int*   targets = (const int*)d_in[2];     // (B,) i32
  float* out = (float*)d_out;
  float* ws  = (float*)d_ws;

  const int B = in_sizes[0] / K_DIM;              // 4096

  hipLaunchKernelGGL(cacis_main_kernel, dim3(B), dim3(256), LDS_BYTES, stream,
                     scores, C, targets, ws);
  hipLaunchKernelGGL(cacis_reduce_kernel, dim3(1), dim3(256), 0, stream,
                     ws, out, B);
  (void)n_in; (void)out_size; (void)ws_size;
}
